// TS_Model2_41283225649718
// MI455X (gfx1250) — compile-verified, matrix operand order checked by reading, not run on hardware
//
#include <hip/hip_runtime.h>

// ---------------- CDNA5 WMMA / TDM types ----------------
typedef __attribute__((ext_vector_type(16))) _Float16 v16h;
typedef __attribute__((ext_vector_type(8)))  float    v8f;
typedef __attribute__((ext_vector_type(4)))  unsigned uint32x4;
typedef __attribute__((ext_vector_type(8)))  int      int32x8;
typedef __attribute__((ext_vector_type(4)))  int      int32x4;

#define BM 32
#define BN 128
#define BK 32

__device__ __forceinline__ int imax(int a, int b) { return a > b ? a : b; }
__device__ __forceinline__ int imin(int a, int b) { return a < b ? a : b; }

// =====================================================================
// Tensor Data Mover: async 2D f16 tile load Global -> LDS.
// D# per CDNA5 ISA (group0: count/lds/global/type=2,
// group1: data_size=1(2B), tensor dims = remaining valid region so OOB
// reads zero-fill the tile, tile dims = BKxBN, dim0 stride = ldw).
// This toolchain exposes the 6-arg builtin form:
//   (uint32x4 g0, int32x8 g1, int32x4 g2, int32x4 g3, int32x8 g4, i32 cpol)
// =====================================================================
__device__ __forceinline__ void tdm_load_tile_f16(
    const _Float16* gptr,   // global address of tile start
    unsigned lds_off,       // LDS byte address of destination
    unsigned t_cols,        // tensor_dim0: valid cols remaining (zero-fill past)
    unsigned t_rows,        // tensor_dim1: valid rows remaining
    unsigned tile_cols,     // tile_dim0
    unsigned tile_rows,     // tile_dim1
    unsigned stride)        // tensor_dim0_stride (elements)
{
    unsigned long long ga = (unsigned long long)gptr;
    uint32x4 g0;
    g0[0] = 1u;                                               // count=1, user desc
    g0[1] = lds_off;                                          // lds_addr
    g0[2] = (unsigned)ga;                                     // global_addr[31:0]
    g0[3] = (unsigned)((ga >> 32) & 0x01FFFFFFu) | (2u << 30);// addr[56:32] | type=2

    int32x8 g1;
    g1[0] = (int)(1u << 16);                                  // data_size=1 (2 bytes)
    g1[1] = (int)((t_cols & 0xFFFFu) << 16);                  // tensor_dim0[15:0] @48
    g1[2] = (int)(((t_cols >> 16) & 0xFFFFu) |
                  ((t_rows & 0xFFFFu) << 16));                // dim0 hi | dim1 lo
    g1[3] = (int)(((t_rows >> 16) & 0xFFFFu) |
                  ((tile_cols & 0xFFFFu) << 16));             // dim1 hi | tile_dim0
    g1[4] = (int)(tile_rows & 0xFFFFu);                       // tile_dim1, tile_dim2=0
    g1[5] = (int)stride;                                      // dim0_stride[31:0]
    g1[6] = 0;                                                // stride hi | dim1_stride lo
    g1[7] = 0;
    int32x4 z4 = {0, 0, 0, 0};
    int32x8 z8 = {0, 0, 0, 0, 0, 0, 0, 0};
    __builtin_amdgcn_tensor_load_to_lds(g0, g1, z4, z4, z8, 0);
}

// =====================================================================
// GEMM: C[M,N] (+)= A[M,K](f32) * W[N,K](f16)^T  (+ bias, optional tanh)
// A converted f32->f16 while staging to LDS (uniform fast path, guarded
// path only on edge tiles). W tile DMA'd by the Tensor Data Mover.
// Block: 256 thr = 8 waves, tile 32(M) x 128(N); each wave owns a 16x32
// strip: one A frag feeds two v_wmma_f32_16x16x32_f16 per K-step.
// =====================================================================
__global__ __launch_bounds__(256)
void gemm_bias_act(const float* __restrict__ A, int lda,
                   const _Float16* __restrict__ W, int ldw,
                   float* __restrict__ C, long long ldc,
                   int M, int N, int K,
                   const float* __restrict__ bias,
                   int act, int accum)
{
    __shared__ _Float16 As[BM * BK];   // 2 KB
    __shared__ _Float16 Ws[BN * BK];   // 8 KB

    const int tid   = threadIdx.x;
    const int m0    = blockIdx.y * BM;
    const int n0    = blockIdx.x * BN;
    const int wave  = tid >> 5;
    const int lane  = tid & 31;
    const int wm    = wave >> 2;   // 0..1  (16-row strip)
    const int wn    = wave & 3;    // 0..3  (32-col strip)
    const int lrow  = lane & 15;
    const int khalf = lane >> 4;

    // C/D fragment mapping (16x16 f32): row=r+(lane>=16?8:0), col=lane&15
    const int ccol0 = n0 + wn * 32 + lrow;
    const int ccol1 = ccol0 + 16;
    int crow[8];
#pragma unroll
    for (int r = 0; r < 8; ++r) crow[r] = m0 + wm * 16 + r + khalf * 8;

    v8f acc0, acc1;
#pragma unroll
    for (int r = 0; r < 8; ++r) { acc0[r] = 0.f; acc1[r] = 0.f; }
    if (accum) {
#pragma unroll
        for (int r = 0; r < 8; ++r) {
            if (crow[r] < M) {
                if (ccol0 < N) acc0[r] = C[(long long)crow[r] * ldc + ccol0];
                if (ccol1 < N) acc1[r] = C[(long long)crow[r] * ldc + ccol1];
            }
        }
    }

    const unsigned ldsW = (unsigned)(unsigned long long)(const void*)Ws;

    for (int k0 = 0; k0 < K; k0 += BK) {
        // ---- W tile via Tensor Data Mover (wave 0 issues; EXEC ignored) ----
        if (tid < 32) {
            tdm_load_tile_f16(W + (long long)n0 * ldw + k0, ldsW,
                              (unsigned)(K - k0), (unsigned)(N - n0),
                              BK, BN, (unsigned)ldw);
        }

        // ---- A tile (BM x BK = 1024 f32->f16, 4 contiguous per thread) ----
        {
            const int r = tid >> 3, c = (tid & 7) * 4;
            if (m0 + BM <= M && k0 + BK <= K) {          // uniform fast path
                const float* src = A + (long long)(m0 + r) * lda + k0 + c;
                As[r * BK + c + 0] = (_Float16)src[0];
                As[r * BK + c + 1] = (_Float16)src[1];
                As[r * BK + c + 2] = (_Float16)src[2];
                As[r * BK + c + 3] = (_Float16)src[3];
            } else {                                     // edge tiles only
#pragma unroll
                for (int i = 0; i < 4; ++i) {
                    float v = 0.f;
                    int gr = m0 + r, gc = k0 + c + i;
                    if (gr < M && gc < K) v = A[(long long)gr * lda + gc];
                    As[r * BK + c + i] = (_Float16)v;
                }
            }
        }
        if (k0 + BK < K && (m0 + (tid >> 3)) < M)
            __builtin_prefetch(&A[(long long)(m0 + (tid >> 3)) * lda + k0 + BK], 0, 1);

        __builtin_amdgcn_s_wait_tensorcnt(0);   // non-issuing waves: cnt==0, no-op
        __syncthreads();

        // ---- fragments from LDS ----
        union { v16h v; unsigned u[8]; } fa, fb0, fb1;
        const unsigned* Au = (const unsigned*)As;
        const unsigned* Wu = (const unsigned*)Ws;
        const int arow  = wm * 16 + lrow;
        const int bcol0 = wn * 32 + lrow;
#pragma unroll
        for (int j = 0; j < 8; ++j) {
            int kk = (j < 4) ? (khalf * 8 + 2 * j) : (16 + khalf * 8 + 2 * (j - 4));
            fa.u[j]  = Au[arow * (BK / 2) + (kk >> 1)];
            fb0.u[j] = Wu[bcol0 * (BK / 2) + (kk >> 1)];
            fb1.u[j] = Wu[(bcol0 + 16) * (BK / 2) + (kk >> 1)];
        }
        acc0 = __builtin_amdgcn_wmma_f32_16x16x32_f16(
                   false, fa.v, false, fb0.v, (short)0, acc0, false, false);
        acc1 = __builtin_amdgcn_wmma_f32_16x16x32_f16(
                   false, fa.v, false, fb1.v, (short)0, acc1, false, false);
        __syncthreads();
    }

    // ---- epilogue ----
#pragma unroll
    for (int r = 0; r < 8; ++r) {
        if (crow[r] >= M) continue;
        if (ccol0 < N) {
            float v = acc0[r];
            if (!accum && bias) v += bias[ccol0];
            if (act == 1) v = tanhf(v);
            C[(long long)crow[r] * ldc + ccol0] = v;
        }
        if (ccol1 < N) {
            float v = acc1[r];
            if (!accum && bias) v += bias[ccol1];
            if (act == 1) v = tanhf(v);
            C[(long long)crow[r] * ldc + ccol1] = v;
        }
    }
}

// =====================================================================
// f32 -> f16 weight conversion (done once per launch, amortized 48x)
// =====================================================================
__global__ void cvt_f16(const float* __restrict__ src, _Float16* __restrict__ dst,
                        long long n)
{
    long long i = (long long)blockIdx.x * blockDim.x + threadIdx.x;
    if (i < n) dst[i] = (_Float16)src[i];
}

// =====================================================================
// LSTM gate update
// =====================================================================
__global__ void lstm_gates(const float* __restrict__ g, long long gStride,
                           float* __restrict__ h, float* __restrict__ c,
                           float* __restrict__ out, long long outStride, int S)
{
    int idx = blockIdx.x * blockDim.x + threadIdx.x;
    if (idx >= S * 128) return;
    int s = idx >> 7, j = idx & 127;
    const float* gr = g + (long long)s * gStride;
    float gi = gr[j], gf = gr[128 + j], gg = gr[256 + j], go = gr[384 + j];
    float i_ = 1.f / (1.f + expf(-gi));
    float f_ = 1.f / (1.f + expf(-gf));
    float o_ = 1.f / (1.f + expf(-go));
    float cc = f_ * c[idx] + i_ * tanhf(gg);
    float hh = o_ * tanhf(cc);
    c[idx] = cc;
    h[idx] = hh;
    out[(long long)s * outStride + j] = hh;
}

__global__ void fill_f32(float* __restrict__ p, long long n, float v)
{
    long long i = (long long)blockIdx.x * blockDim.x + threadIdx.x;
    if (i < n) p[i] = v;
}

// =====================================================================
// Word-level attention pool
// =====================================================================
__global__ __launch_bounds__(256)
void word_attn(const float* __restrict__ ann, const float* __restrict__ wout,
               const float* __restrict__ ctxv, float* __restrict__ enc)
{
    __shared__ float lg[48];
    __shared__ float wgt[48];
    const int s = blockIdx.x;
    const int t = threadIdx.x;
    if (t < 48) {
        const float* a = ann + ((long long)s * 48 + t) * 256;
        float d = 0.f;
        for (int k = 0; k < 256; ++k) d += a[k] * ctxv[k];
        lg[t] = d;
    }
    __syncthreads();
    if (t == 0) {
        float m = -1e30f;
        for (int l = 0; l < 48; ++l) m = fmaxf(m, lg[l]);
        float sum = 0.f;
        for (int l = 0; l < 48; ++l) { float e = expf(lg[l] - m); wgt[l] = e; sum += e; }
        float inv = 1.f / sum;
        for (int l = 0; l < 48; ++l) wgt[l] *= inv;
    }
    __syncthreads();
    float acc = 0.f;
    for (int l = 0; l < 48; ++l)
        acc += wgt[l] * wout[((long long)s * 48 + l) * 256 + t];
    enc[(long long)s * 256 + t] = acc;
}

// =====================================================================
// Windowed cosine-gated attention
// =====================================================================
__device__ __forceinline__ float block_reduce(float v, float* red)
{
    int t = threadIdx.x;
    red[t] = v;
    __syncthreads();
    for (int off = 128; off > 0; off >>= 1) {
        if (t < off) red[t] += red[t + off];
        __syncthreads();
    }
    float r = red[0];
    __syncthreads();
    return r;
}

__global__ __launch_bounds__(256)
void ctx_attn(const float* __restrict__ X, const float* __restrict__ attn_w,
              const float* __restrict__ attn_b, float* __restrict__ out, int N)
{
    __shared__ float red[256];
    __shared__ float sh[16];
    const int p = blockIdx.x;
    const int b = p / N, i = p % N;
    const int d = threadIdx.x;
    const float* Xb = X + (long long)b * N * 256;

    float cv   = Xb[(long long)i * 256 + d];
    float cn   = block_reduce(cv * cv, red);
    float dw2  = block_reduce(cv * attn_w[256 + d], red);
    float cinv = rsqrtf(fmaxf(cn, 1e-30f));

    const int start = imax(i - 3, 0);
    const int vmax  = imin(i + 4, N);
    for (int w = 0; w < 7; ++w) {
        int idx = start + w;
        int row = imin(idx, N - 1);
        float xv = Xb[(long long)row * 256 + d];
        float dc = block_reduce(xv * cv, red);
        float nx = block_reduce(xv * xv, red);
        float d1 = block_reduce(xv * attn_w[d], red);
        if (d == 0) {
            float cosv  = dc * rsqrtf(fmaxf(nx, 1e-30f)) * cinv;
            float gate  = 1.f / (1.f + expf(-cosv));
            float logit = d1 + dw2 + gate * attn_w[512] + attn_b[0];
            sh[w] = (idx < vmax) ? logit : -1e30f;
        }
    }
    __syncthreads();
    if (d == 0) {
        float m = -1e30f;
        for (int w = 0; w < 7; ++w) m = fmaxf(m, sh[w]);
        float ssum = 0.f;
        for (int w = 0; w < 7; ++w) { float e = expf(sh[w] - m); sh[8 + w] = e; ssum += e; }
        float inv = 1.f / ssum;
        for (int w = 0; w < 7; ++w) sh[8 + w] *= inv;
    }
    __syncthreads();
    float acc = 0.f;
    for (int w = 0; w < 7; ++w) {
        int row = imin(start + w, N - 1);
        acc += sh[8 + w] * Xb[(long long)row * 256 + d];
    }
    out[(long long)p * 256 + d] = acc;
}

// =====================================================================
// concat / classifier
// =====================================================================
__global__ void concat2(const float* __restrict__ a, int wa,
                        const float* __restrict__ b, int wb,
                        float* __restrict__ out, long long n)
{
    long long idx = (long long)blockIdx.x * blockDim.x + threadIdx.x;
    if (idx >= n) return;
    int w = wa + wb;
    long long s = idx / w;
    int col = (int)(idx - s * w);
    out[idx] = (col < wa) ? a[s * wa + col] : b[s * (long long)wb + (col - wa)];
}

__global__ void final_linear(const float* __restrict__ d2,
                             const float* __restrict__ w,
                             const float* __restrict__ bias,
                             float* __restrict__ out, int N)
{
    int idx = blockIdx.x * blockDim.x + threadIdx.x;
    int total = 16 * (N - 1) * 2;
    if (idx >= total) return;
    int o = idx & 1, r = idx >> 1;
    int bb = r / (N - 1), n = r % (N - 1);
    const float* src = d2 + ((long long)bb * N + n) * 256;
    float acc = bias[o];
    for (int k = 0; k < 256; ++k) acc += src[k] * w[o * 256 + k];
    out[idx] = acc;
}

// =====================================================================
// Host orchestration
// =====================================================================
extern "C" void kernel_launch(void* const* d_in, const int* in_sizes, int n_in,
                              void* d_out, int out_size, void* d_ws, size_t ws_size,
                              hipStream_t stream)
{
    (void)in_sizes; (void)n_in; (void)out_size; (void)ws_size;
#define IN(i) ((const float*)d_in[i])

    const int B = 16, N = 128, L = 48, E = 300;
    const long long TOK = (long long)B * N * L;   // 98304
    const int S = B * N;                          // 2048

    // ---- workspace carve-up (floats) ----
    float* ws    = (float*)d_ws;
    float* xg    = ws;                          // TOK*512
    float* bufA  = xg   + TOK * 512;            // TOK*256
    float* bufB  = bufA + TOK * 256;            // TOK*256
    float* hbuf  = bufB + TOK * 256;            // S*128
    float* cbuf  = hbuf + (long long)S * 128;
    float* enc   = cbuf + (long long)S * 128;   // S*256
    float* feats = enc  + (long long)S * 256;   // S*1024
    float* doc0  = feats+ (long long)S * 1024;  // S*256
    float* docO  = doc0 + (long long)S * 256;
    float* ctxb  = docO + (long long)S * 256;
    float* d2in  = ctxb + (long long)S * 256;   // S*512
    float* d2a   = d2in + (long long)S * 512;
    float* d2b   = d2a  + (long long)S * 256;

    // f16 weight staging region (after f32 buffers)
    _Float16* f16p = (_Float16*)(d2b + (long long)S * 256);
    auto cvtW = [&](const float* src, long long n) -> const _Float16* {
        _Float16* dst = f16p;
        f16p += n;
        cvt_f16<<<(unsigned)((n + 255) / 256), 256, 0, stream>>>(src, dst, n);
        return dst;
    };

    // convert all GEMM weights once (reused up to 128x in the recurrences)
    const _Float16 *Wih16[12], *Whh16[12];
    const int ihK[12]  = {300, 300, 256, 256, 1024, 1024, 256, 256, 512, 512, 256, 256};
    const int idx0[12] = {2, 5, 8, 11, 17, 20, 23, 26, 29, 32, 35, 38};
    for (int i = 0; i < 12; ++i) {
        Wih16[i] = cvtW(IN(idx0[i]),     512LL * ihK[i]);
        Whh16[i] = cvtW(IN(idx0[i] + 1), 512LL * 128);
    }
    const _Float16* mlp16 = cvtW(IN(14), 256LL * 256);

    auto gemm = [&](const float* A, int lda, const _Float16* W, int ldw,
                    float* C, long long ldc, int M, int Nn, int K,
                    const float* bias, int act, int accum) {
        dim3 grid((Nn + BN - 1) / BN, (M + BM - 1) / BM);
        gemm_bias_act<<<grid, 256, 0, stream>>>(A, lda, W, ldw, C, ldc,
                                                M, Nn, K, bias, act, accum);
    };

    // One BiLSTM direction over tokens stored as [S, T, I] row-major.
    auto run_dir = [&](const float* X, int I, int Sd, int T,
                       const _Float16* Wih, const _Float16* Whh, const float* bias,
                       float* outp, int outW, int colOff, bool rev) {
        gemm(X, I, Wih, I, xg, 512, Sd * T, 512, I, bias, 0, 0);
        long long hn = (long long)Sd * 128;
        unsigned fb = (unsigned)((hn + 255) / 256);
        fill_f32<<<fb, 256, 0, stream>>>(hbuf, hn, 0.f);
        fill_f32<<<fb, 256, 0, stream>>>(cbuf, hn, 0.f);
        for (int st = 0; st < T; ++st) {
            int t = rev ? (T - 1 - st) : st;
            gemm(hbuf, 128, Whh, 128, xg + (long long)t * 512,
                 (long long)T * 512, Sd, 512, 128, nullptr, 0, 1);
            lstm_gates<<<(Sd * 128 + 255) / 256, 256, 0, stream>>>(
                xg + (long long)t * 512, (long long)T * 512, hbuf, cbuf,
                outp + (long long)t * outW + colOff, (long long)T * outW, Sd);
        }
    };

    // ---------------- word-level BiLSTM (2 layers) ----------------
    const float* word_emb = IN(0);
    const float* bert     = IN(1);
    run_dir(word_emb, E, S, L, Wih16[0], Whh16[0], IN(4),  bufA, 256, 0,   false);
    run_dir(word_emb, E, S, L, Wih16[1], Whh16[1], IN(7),  bufA, 256, 128, true);
    run_dir(bufA, 256, S, L, Wih16[2], Whh16[2], IN(10), bufB, 256, 0,   false);
    run_dir(bufA, 256, S, L, Wih16[3], Whh16[3], IN(13), bufB, 256, 128, true);

    // ---------------- sentence encoder attention ----------------
    gemm(bufB, 256, mlp16, 256, xg, 256, (int)TOK, 256, 256, IN(15), 1, 0);
    word_attn<<<S, 256, 0, stream>>>(xg, bufB, IN(16), enc);
    long long fn = (long long)S * 1024;
    concat2<<<(unsigned)((fn + 255) / 256), 256, 0, stream>>>(enc, 256, bert, 768, feats, fn);

    // ---------------- doc-level BiLSTM #1 ----------------
    run_dir(feats, 1024, B, N, Wih16[4], Whh16[4], IN(19), doc0, 256, 0,   false);
    run_dir(feats, 1024, B, N, Wih16[5], Whh16[5], IN(22), doc0, 256, 128, true);
    run_dir(doc0,  256,  B, N, Wih16[6], Whh16[6], IN(25), docO, 256, 0,   false);
    run_dir(doc0,  256,  B, N, Wih16[7], Whh16[7], IN(28), docO, 256, 128, true);

    // ---------------- windowed cosine attention ----------------
    ctx_attn<<<S, 256, 0, stream>>>(docO, IN(41), IN(42), ctxb, N);
    long long dn = (long long)S * 512;
    concat2<<<(unsigned)((dn + 255) / 256), 256, 0, stream>>>(docO, 256, ctxb, 256, d2in, dn);

    // ---------------- doc-level BiLSTM #2 ----------------
    run_dir(d2in, 512, B, N, Wih16[8],  Whh16[8],  IN(31), d2a, 256, 0,   false);
    run_dir(d2in, 512, B, N, Wih16[9],  Whh16[9],  IN(34), d2a, 256, 128, true);
    run_dir(d2a,  256, B, N, Wih16[10], Whh16[10], IN(37), d2b, 256, 0,   false);
    run_dir(d2a,  256, B, N, Wih16[11], Whh16[11], IN(40), d2b, 256, 128, true);

    // ---------------- classifier ----------------
    final_linear<<<(B * (N - 1) * 2 + 255) / 256, 256, 0, stream>>>(
        d2b, IN(43), IN(44), (float*)d_out, N);
#undef IN
}
